// OnnxDistanceEstimatorWrapper_54700703481926
// MI455X (gfx1250) — matrix operand order repaired and checked by reading, not compiled, not run on hardware
//
#include <hip/hip_runtime.h>
#include <hip/hip_bf16.h>

// CDNA5 / gfx1250, wave32. GEMMs use v_wmma_f32_16x16x32_f16; weight tiles are
// staged into LDS by the Tensor Data Mover (tensor_load_to_lds + s_wait_tensorcnt).

typedef __attribute__((ext_vector_type(16))) _Float16 v16h;
typedef __attribute__((ext_vector_type(8)))  _Float16 v8h;
typedef __attribute__((ext_vector_type(8)))  float    v8f;
typedef __attribute__((ext_vector_type(4)))  float    v4f;
typedef __attribute__((ext_vector_type(4)))  unsigned int v4u;
typedef __attribute__((ext_vector_type(8)))  int      v8i;
typedef __attribute__((ext_vector_type(4)))  int      v4i;

#define HDIM 128

#if __has_builtin(__builtin_amdgcn_tensor_load_to_lds)
#define USE_TDM 1
#else
#define USE_TDM 0
#endif

// ---------------------------------------------------------------- zero fill
__global__ void zero_f32(float* __restrict__ p, long n) {
  long i4 = ((long)blockIdx.x * blockDim.x + threadIdx.x) * 4;
  if (i4 + 3 < n) {
    *(v4f*)(p + i4) = (v4f){0.f, 0.f, 0.f, 0.f};
  } else {
    for (long j = i4; j < n; ++j) p[j] = 0.f;
  }
}

// ------------------------------------------- weights: f32 [k][n] -> f16 Wt[n][k]
// Pre-transposed into WMMA B-operand friendly order: lane n reads contiguous K.
__global__ void convert_wt(const float* w0, const float* w1, const float* w2,
                           const float* w3, const float* w4, const float* w5,
                           _Float16* __restrict__ wt) {
  const float* srcs[6] = {w0, w1, w2, w3, w4, w5};
  int mat = blockIdx.y;
  int t = blockIdx.x * blockDim.x + threadIdx.x;   // 0 .. 128*128-1
  int n = t / HDIM, k = t % HDIM;
  wt[(size_t)mat * HDIM * HDIM + n * HDIM + k] = (_Float16)srcs[mat][k * HDIM + n];
}

// ------------------------------------------------- id_mlp layer 1 (K = 1)
__global__ void id_l1(const int* __restrict__ ids, const float* __restrict__ w1,
                      const float* __restrict__ b1, _Float16* __restrict__ out, int N_) {
  long t = (long)blockIdx.x * blockDim.x + threadIdx.x;
  if (t >= (long)N_ * HDIM) return;
  int n = (int)(t / HDIM), h = (int)(t % HDIM);
  float x0 = (float)ids[n] * (1.0f / 281474976710655.0f);   // / (2^48 - 1)
  x0 = fminf(fmaxf(x0, 0.f), 1.f);
  float v = x0 * w1[h] + b1[h];
  out[t] = (_Float16)fmaxf(v, 0.f);
}

// ----------------------------------------------- edge_mlp layer 1 (K = ED <= 8)
__global__ void edge_l1(const float* __restrict__ attr, const float* __restrict__ w1,
                        const float* __restrict__ b1, _Float16* __restrict__ out,
                        int E_, int ED_) {
  __shared__ float w1s[8 * HDIM];
  __shared__ float b1s[HDIM];
  int h = threadIdx.x;                      // 128 threads
  int edc = ED_ < 8 ? ED_ : 8;
  for (int k = 0; k < edc; ++k) w1s[k * HDIM + h] = w1[k * HDIM + h];
  b1s[h] = b1[h];
  __syncthreads();
  int e0 = blockIdx.x * 32;
  for (int j = 0; j < 32; ++j) {
    int e = e0 + j;
    if (e >= E_) break;
    float acc = b1s[h];
    #pragma unroll 8
    for (int k = 0; k < edc; ++k) acc += attr[(long)e * ED_ + k] * w1s[k * HDIM + h];
    out[(long)e * HDIM + h] = (_Float16)fmaxf(acc, 0.f);
  }
}

// --------------------------------------------------------- WMMA GEMM [M,128]x[128,128]
// One block = 8 waves = 128 rows. Full Wt (f16, B-operand order) staged into LDS
// (32KB of 320KB/WGP) by the TDM. FUSED_ADD: A = f16(f32 Aadd + f16 A16) (GINE
// x + aggr fusion). RELU/bias fused into the epilogue. Output f16.
template <bool FUSED_ADD, bool RELU>
__global__ __launch_bounds__(256) void gemm_h128(
    const _Float16* __restrict__ A16, const float* __restrict__ Aadd,
    const _Float16* __restrict__ WtG, const float* __restrict__ bias,
    _Float16* __restrict__ out16, int M) {
  __shared__ _Float16 Wt[HDIM * HDIM];          // 32 KB

#if USE_TDM
  // Tensor Data Mover: one wave DMAs the 2D 128x128 f16 tile (stride 128) into
  // LDS offset 0. TENSORcnt tracks completion; barrier publishes to the block.
  if (threadIdx.x < 32) {
    unsigned long long ga = (unsigned long long)(uintptr_t)WtG;
    v4u g0 = { 1u,                                        // count=1 valid descriptor
               0u,                                        // lds_addr = 0 (Wt base)
               (unsigned)(ga & 0xFFFFFFFFu),              // global_addr[31:0]
               (unsigned)((ga >> 32) & 0x01FFFFFFu) | (2u << 30) }; // addr[56:32] | type=2
    v8i g1 = { (int)(1u << 16),                           // wg_mask=0, data_size=1 (2B)
               (int)(128u << 16),                         // tensor_dim0 lo16 = 128
               (int)(128u << 16),                         // dim0 hi=0 | tensor_dim1 lo16=128
               (int)(128u << 16),                         // dim1 hi=0 | tile_dim0=128
               (int)(128u),                               // tile_dim1=128, tile_dim2=0
               (int)(128u),                               // tensor_dim0_stride = 128
               (int)(16384u << 16),                       // d0_stride hi=0 | d1_stride lo16
               0 };                                       // d1_stride hi (16384>>16 = 0)
    v4i g2 = {0, 0, 0, 0};
    v4i g3 = {0, 0, 0, 0};
    v8i g4 = {0, 0, 0, 0, 0, 0, 0, 0};                    // unused group (6-arg lane)
    __builtin_amdgcn_tensor_load_to_lds(g0, g1, g2, g3, g4, 0);
    __builtin_amdgcn_s_wait_tensorcnt(0);
  }
  __syncthreads();
#else
  {   // fallback: cooperative 16B-chunk copy
    const uint4* src = (const uint4*)WtG;
    uint4* dst = (uint4*)Wt;
    #pragma unroll
    for (int i = 0; i < 8; ++i) dst[threadIdx.x + i * 256] = src[threadIdx.x + i * 256];
  }
  __syncthreads();
#endif

  const int lane = threadIdx.x & 31;
  const int wave = threadIdx.x >> 5;
  const int m16  = lane & 15;      // A row within tile / B,C,D column
  const int kh   = lane >> 4;      // K half-select (ISA 7.12.2 wave32 layouts)
  const int row0 = blockIdx.x * 128 + wave * 16;
  const bool fullTile = (row0 + 16) <= M;   // wave-uniform: single s_cbranch
  int rowA = row0 + m16;
  if (rowA >= M) rowA = M - 1;     // clamp; stores are guarded in the tail path

  // A fragments for the 4 K-steps (K = 128 = 4 x 32)
  v16h afrag[4];
  #pragma unroll
  for (int ks = 0; ks < 4; ++ks) {
    const int kb = ks * 32 + kh * 8;
    if (FUSED_ADD) {
      const float*    pa = Aadd + (long)rowA * HDIM;
      const _Float16* px = A16  + (long)rowA * HDIM;
      #pragma unroll
      for (int half = 0; half < 2; ++half) {
        const int off = kb + half * 16;
        v4f f0 = *(const v4f*)(pa + off);
        v4f f1 = *(const v4f*)(pa + off + 4);
        v8h x8 = *(const v8h*)(px + off);
        #pragma unroll
        for (int i = 0; i < 4; ++i) {
          afrag[ks][half * 8 + i]     = (_Float16)(f0[i] + (float)x8[i]);
          afrag[ks][half * 8 + 4 + i] = (_Float16)(f1[i] + (float)x8[4 + i]);
        }
      }
    } else {
      const _Float16* px = A16 + (long)rowA * HDIM;
      v8h lo = *(const v8h*)(px + kb);        // K = kb..kb+7     -> VGPR 0..3
      v8h hi = *(const v8h*)(px + kb + 16);   // K = kb+16..kb+23 -> VGPR 4..7
      #pragma unroll
      for (int i = 0; i < 8; ++i) { afrag[ks][i] = lo[i]; afrag[ks][8 + i] = hi[i]; }
    }
  }

  // 8 N-tiles of 16 columns; accumulator live one tile at a time
  #pragma unroll
  for (int nt = 0; nt < 8; ++nt) {
    v8f acc = (v8f){0.f, 0.f, 0.f, 0.f, 0.f, 0.f, 0.f, 0.f};
    #pragma unroll
    for (int ks = 0; ks < 4; ++ks) {
      // B operand: lane -> column n = lane%16, K = (lane/16)*8 + 0..7 and +16 + 0..7
      const _Float16* pb = Wt + (nt * 16 + m16) * HDIM + ks * 32 + kh * 8;
      v8h blo = *(const v8h*)pb;
      v8h bhi = *(const v8h*)(pb + 16);
      v16h bfrag;
      #pragma unroll
      for (int i = 0; i < 8; ++i) { bfrag[i] = blo[i]; bfrag[8 + i] = bhi[i]; }
      acc = __builtin_amdgcn_wmma_f32_16x16x32_f16(
                false, afrag[ks], false, bfrag, (short)0, acc, false, false);
    }
    // Epilogue. C/D layout: col = lane%16; row = i + 8*(lane>=16)
    const int col = nt * 16 + m16;
    const float bv = bias[col];
    _Float16* po = out16 + (long)(row0 + kh * 8) * HDIM + col;
    if (fullTile) {
      // fast path: straight-line stores, immediate row offsets
      #pragma unroll
      for (int i = 0; i < 8; ++i) {
        float v = acc[i] + bv;
        if (RELU) v = fmaxf(v, 0.f);
        po[(long)i * HDIM] = (_Float16)v;
      }
    } else {
      const int mbase = row0 + kh * 8;
      #pragma unroll
      for (int i = 0; i < 8; ++i) {
        if (mbase + i < M) {
          float v = acc[i] + bv;
          if (RELU) v = fmaxf(v, 0.f);
          po[(long)i * HDIM] = (_Float16)v;
        }
      }
    }
  }
}

// ---------------------------------------- GINE message + segment-sum scatter
// msg = relu(x[src] + e), f32 atomic accumulate at dst. 8 channels per thread.
__global__ void scatter_msgs(const int* __restrict__ ei, const _Float16* __restrict__ x16,
                             const _Float16* __restrict__ e16, float* __restrict__ aggr,
                             int E_) {
  long t = (long)blockIdx.x * blockDim.x + threadIdx.x;
  if (t >= (long)E_ * 16) return;
  int e  = (int)(t >> 4);
  int h0 = ((int)t & 15) * 8;
  int src = ei[e], dst = ei[E_ + e];
  v8h xs = *(const v8h*)(x16 + (long)src * HDIM + h0);
  v8h es = *(const v8h*)(e16 + (long)e   * HDIM + h0);
  float* dp = aggr + (long)dst * HDIM + h0;
  #pragma unroll
  for (int i = 0; i < 8; ++i) {
    float m = fmaxf((float)xs[i] + (float)es[i], 0.f);
    atomicAdd(dp + i, m);
  }
}

// ---------------------------------------------------- global mean pool (sums)
__global__ void pool_sum(const int* __restrict__ batch, const _Float16* __restrict__ x16,
                         float* __restrict__ sums, float* __restrict__ cnt, int N_) {
  long t = (long)blockIdx.x * blockDim.x + threadIdx.x;
  if (t >= (long)N_ * 16) return;
  int n  = (int)(t >> 4);
  int h0 = ((int)t & 15) * 8;
  int b = batch[n];
  v8h xs = *(const v8h*)(x16 + (long)n * HDIM + h0);
  float* dp = sums + (long)b * HDIM + h0;
  #pragma unroll
  for (int i = 0; i < 8; ++i) atomicAdd(dp + i, (float)xs[i]);
  if ((t & 15) == 0) atomicAdd(cnt + b, 1.0f);
}

// --------------------------------------------- regressor: [mean,depth] MLP -> scalar
__global__ void regressor(const float* __restrict__ sums, const float* __restrict__ cnt,
                          const float* __restrict__ depth, const float* __restrict__ w1,
                          const float* __restrict__ b1, const float* __restrict__ w2,
                          const float* __restrict__ b2, float* __restrict__ out) {
  int b = blockIdx.x;
  int h = threadIdx.x;            // 128 threads
  __shared__ float mean[HDIM];
  __shared__ float red[HDIM];
  float c = fmaxf(cnt[b], 1.0f);
  mean[h] = sums[(long)b * HDIM + h] / c;
  __syncthreads();
  float acc = b1[h];
  #pragma unroll 4
  for (int k = 0; k < HDIM; ++k) acc += mean[k] * w1[k * HDIM + h];
  acc += depth[b] * w1[HDIM * HDIM + h];      // w1 is [H+1, H]; depth is row H
  acc = fmaxf(acc, 0.f);
  red[h] = acc * w2[h];                       // w2 is [H,1]
  __syncthreads();
  for (int s = 64; s > 0; s >>= 1) {
    if (h < s) red[h] += red[h + s];
    __syncthreads();
  }
  if (h == 0) out[b] = red[0] + b2[0];
}

// ================================================================ launch
extern "C" void kernel_launch(void* const* d_in, const int* in_sizes, int n_in,
                              void* d_out, int out_size, void* d_ws, size_t ws_size,
                              hipStream_t stream) {
  (void)n_in; (void)out_size; (void)ws_size;
  const int*   ids   = (const int*)d_in[0];
  const int*   ei    = (const int*)d_in[1];
  const float* attr  = (const float*)d_in[2];
  const int*   batch = (const int*)d_in[3];
  const float* depth = (const float*)d_in[4];
  const float* id_w1 = (const float*)d_in[5];
  const float* id_b1 = (const float*)d_in[6];
  const float* id_w2 = (const float*)d_in[7];
  const float* id_b2 = (const float*)d_in[8];
  const float* ed_w1 = (const float*)d_in[9];
  const float* ed_b1 = (const float*)d_in[10];
  const float* ed_w2 = (const float*)d_in[11];
  const float* ed_b2 = (const float*)d_in[12];
  const float* c1_w1 = (const float*)d_in[13];
  const float* c1_b1 = (const float*)d_in[14];
  const float* c1_w2 = (const float*)d_in[15];
  const float* c1_b2 = (const float*)d_in[16];
  const float* c2_w1 = (const float*)d_in[17];
  const float* c2_b1 = (const float*)d_in[18];
  const float* c2_w2 = (const float*)d_in[19];
  const float* c2_b2 = (const float*)d_in[20];
  const float* rg_w1 = (const float*)d_in[21];
  const float* rg_b1 = (const float*)d_in[22];
  const float* rg_w2 = (const float*)d_in[23];
  const float* rg_b2 = (const float*)d_in[24];

  const int N_  = in_sizes[0];
  const int E_  = in_sizes[1] / 2;
  const int B_  = in_sizes[4];
  const int ED_ = in_sizes[2] / E_;

  // -------- workspace carving (256B aligned)
  char* p = (char*)d_ws;
  auto carve = [&](size_t bytes) -> void* {
    void* r = (void*)p;
    p += (bytes + 255) & ~(size_t)255;
    return r;
  };
  _Float16* tmp16 = (_Float16*)carve((size_t)E_ * HDIM * 2);  // MLP hiddens (E >= N)
  _Float16* e16   = (_Float16*)carve((size_t)E_ * HDIM * 2);  // edge embeddings
  _Float16* xA    = (_Float16*)carve((size_t)N_ * HDIM * 2);  // node feats (ping)
  _Float16* xB    = (_Float16*)carve((size_t)N_ * HDIM * 2);  // node feats (pong)
  float*    aggr  = (float*)   carve((size_t)N_ * HDIM * 4);  // segment sums
  _Float16* wt    = (_Float16*)carve((size_t)6 * HDIM * HDIM * 2); // f16 B-op weights
  float*    sums  = (float*)   carve((size_t)B_ * HDIM * 4);
  float*    cnt   = (float*)   carve((size_t)B_ * 4);

  const int TB = 256;
  auto blocksFor = [](long work, int tb) { return (unsigned)((work + tb - 1) / tb); };
  const unsigned gN   = (unsigned)((N_ + 127) / 128);
  const unsigned gE   = (unsigned)((E_ + 127) / 128);
  const unsigned gZa  = blocksFor(((long)N_ * HDIM) / 4, TB);
  const unsigned gSc  = blocksFor((long)E_ * 16, TB);
  const unsigned gPo  = blocksFor((long)N_ * 16, TB);

  // 0) weights -> f16, transposed to B-operand order
  convert_wt<<<dim3(HDIM * HDIM / TB, 6), TB, 0, stream>>>(
      id_w2, ed_w2, c1_w1, c1_w2, c2_w1, c2_w2, wt);
  _Float16* wt_id2 = wt + 0 * HDIM * HDIM;
  _Float16* wt_ed2 = wt + 1 * HDIM * HDIM;
  _Float16* wt_c11 = wt + 2 * HDIM * HDIM;
  _Float16* wt_c12 = wt + 3 * HDIM * HDIM;
  _Float16* wt_c21 = wt + 4 * HDIM * HDIM;
  _Float16* wt_c22 = wt + 5 * HDIM * HDIM;

  // 1) node embedding: id_mlp
  id_l1<<<blocksFor((long)N_ * HDIM, TB), TB, 0, stream>>>(ids, id_w1, id_b1, tmp16, N_);
  gemm_h128<false, false><<<gN, TB, 0, stream>>>(tmp16, nullptr, wt_id2, id_b2, xA, N_);

  // 2) edge embedding: edge_mlp
  edge_l1<<<(unsigned)((E_ + 31) / 32), HDIM, 0, stream>>>(attr, ed_w1, ed_b1, tmp16, E_, ED_);
  gemm_h128<false, false><<<gE, TB, 0, stream>>>(tmp16, nullptr, wt_ed2, ed_b2, e16, E_);

  // 3) GINE conv 1: aggr = segsum(relu(x[src]+e)); x = relu(mlp2(x + aggr))
  zero_f32<<<gZa, TB, 0, stream>>>(aggr, (long)N_ * HDIM);
  scatter_msgs<<<gSc, TB, 0, stream>>>(ei, xA, e16, aggr, E_);
  gemm_h128<true,  true><<<gN, TB, 0, stream>>>(xA, aggr, wt_c11, c1_b1, tmp16, N_);
  gemm_h128<false, true><<<gN, TB, 0, stream>>>(tmp16, nullptr, wt_c12, c1_b2, xB, N_);

  // 4) GINE conv 2
  zero_f32<<<gZa, TB, 0, stream>>>(aggr, (long)N_ * HDIM);
  scatter_msgs<<<gSc, TB, 0, stream>>>(ei, xB, e16, aggr, E_);
  gemm_h128<true,  true><<<gN, TB, 0, stream>>>(xB, aggr, wt_c21, c2_b1, tmp16, N_);
  gemm_h128<false, true><<<gN, TB, 0, stream>>>(tmp16, nullptr, wt_c22, c2_b2, xA, N_);

  // 5) global mean pool + regressor
  zero_f32<<<blocksFor(((long)B_ * HDIM) / 4, TB), TB, 0, stream>>>(sums, (long)B_ * HDIM);
  zero_f32<<<blocksFor(((long)B_ + 3) / 4, TB), TB, 0, stream>>>(cnt, (long)B_);
  pool_sum<<<gPo, TB, 0, stream>>>(batch, xA, sums, cnt, N_);
  regressor<<<(unsigned)B_, HDIM, 0, stream>>>(sums, cnt, depth, rg_w1, rg_b1,
                                               rg_w2, rg_b2, (float*)d_out);
}
